// EdgePredModel_36112085025196
// MI455X (gfx1250) — matrix operand order repaired and checked by reading, not compiled, not run on hardware
//
#include <hip/hip_runtime.h>
#include <hip/hip_fp16.h>

typedef __attribute__((ext_vector_type(16))) _Float16 v16h;
typedef __attribute__((ext_vector_type(8)))  _Float16 v8h;
typedef __attribute__((ext_vector_type(8)))  float    v8f;

#define N_NODES 100000
#define N_EDGES 1600000
#define IN_F 32
#define OUT_F 64

// Native CDNA5 V_TANH_F32 (single TRANS op, co-executes with WMMA) when available.
#if defined(__has_builtin)
#if __has_builtin(__builtin_amdgcn_tanhf)
#define FAST_TANH(x) __builtin_amdgcn_tanhf(x)
#endif
#endif
#ifndef FAST_TANH
#define FAST_TANH(x) tanhf(x)
#endif

__device__ __forceinline__ float fast_sigmoid(float t) {
    // sigmoid(t) = 0.5 + 0.5*tanh(t/2) -> one v_tanh_f32 instead of exp+rcp branchy path
    return 0.5f + 0.5f * FAST_TANH(0.5f * t);
}

// ---------------- utility kernels ----------------

__global__ void zero_kernel(float* p, int n) {
    int i = blockIdx.x * blockDim.x + threadIdx.x;
    if (i < n) p[i] = 0.0f;
}

// Pre-transpose weights to N-major f16 so WMMA B-fragments are one contiguous
// 32B load per lane (this code's B order, matched by A: lane holds col n=lane&15, 16 contiguous K).
__global__ void prep_kernel(const float* __restrict__ W_l, const float* __restrict__ W_r,
                            const float* __restrict__ W1,
                            _Float16* __restrict__ Wlrt,  // [64][64]  (n-major, k = [W_l;W_r])
                            _Float16* __restrict__ W1t) { // [64][128] (n-major)
    int i = blockIdx.x * blockDim.x + threadIdx.x;
    if (i < 64 * 128) {
        int n = i >> 7, k = i & 127;
        W1t[i] = (_Float16)W1[k * 64 + n];
    }
    int j = i - 64 * 128;
    if (j >= 0 && j < 64 * 64) {
        int n = j >> 6, k = j & 63;
        float v = (k < 32) ? W_l[k * 64 + n] : W_r[(k - 32) * 64 + n];
        Wlrt[j] = (_Float16)v;
    }
}

// Mean-aggregation scatter: native f32 L2 atomics.
__global__ void scatter_kernel(const long long* __restrict__ ei,
                               const float* __restrict__ x,
                               float* __restrict__ agg, float* __restrict__ cnt) {
    int e = blockIdx.x * blockDim.x + threadIdx.x;
    if (e >= N_EDGES) return;
    int s = (int)ei[e];
    int d = (int)ei[N_EDGES + e];
    const float* xr = x + (size_t)s * IN_F;
    float* ag = agg + (size_t)d * IN_F;
#pragma unroll
    for (int k = 0; k < IN_F; ++k) unsafeAtomicAdd(&ag[k], xr[k]);
    unsafeAtomicAdd(&cnt[d], 1.0f);
}

// ---------------- node update: h = tanh([mean|x] @ [W_l;W_r] + b_l) ----------------
// One wave per 16-node tile. A = 16x32 f16 chunks (K=64 total), B from Wlrt.
__global__ void node_kernel(const float* __restrict__ x,
                            const float* __restrict__ agg,
                            const float* __restrict__ cnt,
                            const _Float16* __restrict__ Wlrt,
                            const float* __restrict__ b_l,
                            _Float16* __restrict__ h) {
    int lane = threadIdx.x & 31;
    int wave = threadIdx.x >> 5;
    int tile = blockIdx.x * (blockDim.x >> 5) + wave;
    const int ntiles = N_NODES / 16;  // 6250, exact
    if (tile >= ntiles) return;

    int half = lane >> 4;
    int col  = lane & 15;
    int node = tile * 16 + col;   // A-matrix row owned by this lane
    int o1   = 16 * half;

    float inv = 1.0f / fmaxf(cnt[node], 1.0f);
    const float* ag = agg + (size_t)node * IN_F;
    const float* xr = x   + (size_t)node * IN_F;

    union { v16h v; v8h h8[2]; } a0, a1;
#pragma unroll
    for (int i = 0; i < 8; ++i) {
        a0.h8[0][i] = (_Float16)(ag[o1 + i] * inv);
        a0.h8[1][i] = (_Float16)(ag[8 + o1 + i] * inv);
        a1.h8[0][i] = (_Float16)xr[o1 + i];
        a1.h8[1][i] = (_Float16)xr[8 + o1 + i];
    }

#pragma unroll
    for (int nt = 0; nt < 4; ++nt) {
        int n = nt * 16 + col;
        v16h b0 = *(const v16h*)(Wlrt + n * 64 + 16 * half);       // K chunk 0 (mean)
        v16h b1 = *(const v16h*)(Wlrt + n * 64 + 32 + 16 * half);  // K chunk 1 (x)
        v8f c = {};
        c = __builtin_amdgcn_wmma_f32_16x16x32_f16(false, a0.v, false, b0, (short)0, c, false, false);
        c = __builtin_amdgcn_wmma_f32_16x16x32_f16(false, a1.v, false, b1, (short)0, c, false, false);
        float bb = b_l[n];
#pragma unroll
        for (int r = 0; r < 8; ++r) {
            int m  = r + 8 * half;
            int nd = tile * 16 + m;
            h[(size_t)nd * 64 + n] = (_Float16)FAST_TANH(c[r] + bb);
        }
    }
}

// ---------------- edge MLP: sigmoid(tanh(tanh([h_s|h_d]@W1+b1)@W2+b2)) ----------------
// One wave per 16-edge tile, grid-stride. W1 fragments register-resident (128 VGPRs),
// amortized across the tile loop. 16 v_wmma_f32_16x16x32_f16 per tile.
__global__ void edge_kernel(const long long* __restrict__ ei,
                            const _Float16* __restrict__ h,
                            const _Float16* __restrict__ W1t,
                            const float* __restrict__ b1,
                            const float* __restrict__ W2,
                            const float* __restrict__ b2,
                            float* __restrict__ out) {
    int lane = threadIdx.x & 31;
    int wave = threadIdx.x >> 5;
    int gw   = blockIdx.x * (blockDim.x >> 5) + wave;
    int nw   = gridDim.x * (blockDim.x >> 5);
    int half = lane >> 4;
    int col  = lane & 15;

    // Preload all 16 B fragments of W1 (K=128 -> 4 chunks, N=64 -> 4 tiles).
    v16h B[4][4];
#pragma unroll
    for (int kc = 0; kc < 4; ++kc)
#pragma unroll
        for (int nt = 0; nt < 4; ++nt) {
            int n = nt * 16 + col;
            B[kc][nt] = *(const v16h*)(W1t + n * 128 + kc * 32 + 16 * half);
        }
    float b1v[4], w2v[4];
#pragma unroll
    for (int nt = 0; nt < 4; ++nt) {
        int n = nt * 16 + col;
        b1v[nt] = b1[n];
        w2v[nt] = W2[n];
    }
    float bias2 = b2[0];

    const int ntiles = N_EDGES / 16;  // 100000, exact
    int o1 = 16 * half;

    for (int tile = gw; tile < ntiles; tile += nw) {
        int e = tile * 16 + col;
        int s = (int)ei[e];
        int d = (int)ei[N_EDGES + e];
        const _Float16* hs = h + (size_t)s * 64;
        const _Float16* hd = h + (size_t)d * 64;

        // A: 16x128 he-tile, 4 chunks of 16x32; rows gathered through L2-resident h.
        union { v16h v; v8h h8[2]; } A[4];
        A[0].h8[0] = *(const v8h*)(hs + o1);       A[0].h8[1] = *(const v8h*)(hs + 8 + o1);
        A[1].h8[0] = *(const v8h*)(hs + 32 + o1);  A[1].h8[1] = *(const v8h*)(hs + 40 + o1);
        A[2].h8[0] = *(const v8h*)(hd + o1);       A[2].h8[1] = *(const v8h*)(hd + 8 + o1);
        A[3].h8[0] = *(const v8h*)(hd + 32 + o1);  A[3].h8[1] = *(const v8h*)(hd + 40 + o1);

        float accr[8];
#pragma unroll
        for (int r = 0; r < 8; ++r) accr[r] = 0.0f;

#pragma unroll
        for (int nt = 0; nt < 4; ++nt) {
            v8f c = {};
#pragma unroll
            for (int kc = 0; kc < 4; ++kc)
                c = __builtin_amdgcn_wmma_f32_16x16x32_f16(false, A[kc].v, false, B[kc][nt],
                                                           (short)0, c, false, false);
            // epilogue: +b1, tanh (native v_tanh_f32), scale by W2 column weight
#pragma unroll
            for (int r = 0; r < 8; ++r)
                accr[r] += FAST_TANH(c[r] + b1v[nt]) * w2v[nt];
        }

        // Reduce over the 64 output columns: butterfly across the 16-lane half.
#pragma unroll
        for (int r = 0; r < 8; ++r) {
            float v = accr[r];
            v += __shfl_xor(v, 1, 32);
            v += __shfl_xor(v, 2, 32);
            v += __shfl_xor(v, 4, 32);
            v += __shfl_xor(v, 8, 32);
            accr[r] = v;
        }
        if (col == 0) {
#pragma unroll
            for (int r = 0; r < 8; ++r) {
                int m = r + 8 * half;           // rows 0-7 (half 0) / 8-15 (half 1)
                out[tile * 16 + m] = fast_sigmoid(FAST_TANH(accr[r] + bias2));
            }
        }
    }
}

// ---------------- host launcher ----------------

extern "C" void kernel_launch(void* const* d_in, const int* in_sizes, int n_in,
                              void* d_out, int out_size, void* d_ws, size_t ws_size,
                              hipStream_t stream) {
    const float*     x   = (const float*)d_in[0];
    const long long* ei  = (const long long*)d_in[1];
    const float*     W_l = (const float*)d_in[2];
    const float*     W_r = (const float*)d_in[3];
    const float*     b_l = (const float*)d_in[4];
    const float*     W1  = (const float*)d_in[5];
    const float*     b1  = (const float*)d_in[6];
    const float*     W2  = (const float*)d_in[7];
    const float*     b2  = (const float*)d_in[8];
    float* out = (float*)d_out;

    char* ws = (char*)d_ws;
    // workspace layout (all 16B aligned)
    float*     agg  = (float*)ws;                                           // N*32 f32 = 12.8 MB
    float*     cnt  = (float*)(ws + (size_t)N_NODES * IN_F * 4);            // N f32    = 0.4 MB
    _Float16*  h    = (_Float16*)(ws + (size_t)N_NODES * IN_F * 4
                                     + (size_t)N_NODES * 4);                // N*64 f16 = 12.8 MB
    _Float16*  W1t  = (_Float16*)(ws + 26000000);                           // 64*128 f16
    _Float16*  Wlrt = (_Float16*)(ws + 26000000 + 64 * 128 * 2);            // 64*64 f16

    // 1) zero agg+cnt (contiguous)
    int ztot = N_NODES * IN_F + N_NODES;
    zero_kernel<<<(ztot + 255) / 256, 256, 0, stream>>>(agg, ztot);

    // 2) weight transpose/convert (tiny)
    prep_kernel<<<(64 * 128 + 64 * 64 + 255) / 256, 256, 0, stream>>>(W_l, W_r, W1, Wlrt, W1t);

    // 3) mean-aggregation scatter
    scatter_kernel<<<(N_EDGES + 255) / 256, 256, 0, stream>>>(ei, x, agg, cnt);

    // 4) node update (WMMA, 6250 tiles, 8 waves/block)
    node_kernel<<<(N_NODES / 16 + 7) / 8, 256, 0, stream>>>(x, agg, cnt, Wlrt, b_l, h);

    // 5) edge MLP (WMMA, 100000 tiles, grid-stride persistent waves)
    edge_kernel<<<1664, 256, 0, stream>>>(ei, h, W1t, b1, W2, b2, out);
}
